// DEQSequenceModel_53584011985726
// MI455X (gfx1250) — matrix operand order repaired and checked by reading, not compiled, not run on hardware
//
#include <hip/hip_runtime.h>

// ---------------------------------------------------------------------------
// DEQ sequence model on MI455X (gfx1250):
//   ux  = x @ Wx^T                         (bf16 WMMA, f32 accum)
//   z1  = tanh(bz + ux)                    (elementwise)
//   z_{i+1} = tanh(z_i @ Wz^T + bz + ux)   (bf16 WMMA, fused tanh epilogue) x29
//   out = z30 @ Wd^T + bd                  (bf16 WMMA, f32 out)
// ~1.07 TFLOP total, working set (~90MB) is L2-resident -> matrix-unit bound.
// 64x64 per-wave tile: 16 WMMA per K-step fed by 16 pipelined ds_load_b128
// (all 8 fragments preloaded -> only partial dscnt waits), double-buffered
// LDS software pipeline hides the L2->LDS staging under WMMA work.
// ---------------------------------------------------------------------------

typedef __bf16 bf16;
typedef __attribute__((ext_vector_type(16))) __bf16 v16bf;
typedef __attribute__((ext_vector_type(8)))  __bf16 v8bf;
typedef __attribute__((ext_vector_type(4)))  __bf16 v4bf;
typedef __attribute__((ext_vector_type(8)))  float  v8f;

#define BM 256
#define BN 128
#define BK 32
#define LDSK 40          // BK + 8 elements (16B pad) -> conflict-free ds_load_b128
#define NTHREADS 256

__device__ __forceinline__ bf16 f2bf(float f) {
  // round-to-nearest-even fp32 -> bf16
  unsigned u = __float_as_uint(f);
  unsigned r = (u + 0x7fffu + ((u >> 16) & 1u)) >> 16;
  return __builtin_bit_cast(bf16, (unsigned short)r);
}

// ---------------- fp32 -> bf16 bulk convert (n multiple of 4) --------------
__global__ void cvt_f32_bf16_kernel(const float* __restrict__ s,
                                    bf16* __restrict__ d, int n4) {
  int i = blockIdx.x * blockDim.x + threadIdx.x;
  if (i < n4) {
    float4 f = ((const float4*)s)[i];
    v4bf o = { f2bf(f.x), f2bf(f.y), f2bf(f.z), f2bf(f.w) };
    ((v4bf*)d)[i] = o;
  }
}

// ---------------- z1 = tanh(bz + ux), stored bf16 --------------------------
__global__ void init_z_kernel(const float* __restrict__ ux,
                              const float* __restrict__ bz,
                              bf16* __restrict__ z, int H, int n4) {
  int i = blockIdx.x * blockDim.x + threadIdx.x;
  if (i < n4) {
    int e = i * 4;
    int h = e % H;                       // multiple of 4
    float4 u = ((const float4*)ux)[i];
    const float4 b = *(const float4*)(bz + h);
    v4bf o = { f2bf(tanhf(u.x + b.x)), f2bf(tanhf(u.y + b.y)),
               f2bf(tanhf(u.z + b.z)), f2bf(tanhf(u.w + b.w)) };
    ((v4bf*)z)[i] = o;
  }
}

// ---------------- LDS fragment fetch helper --------------------------------
__device__ __forceinline__ v16bf frag16(const bf16* p0, const bf16* p1) {
  v8bf lo = *(const v8bf*)p0;
  v8bf hi = *(const v8bf*)p1;
  return __builtin_shufflevector(lo, hi, 0,1,2,3,4,5,6,7,8,9,10,11,12,13,14,15);
}

// ---------------------------------------------------------------------------
// Tiled WMMA GEMM:  D[M,N] = A[M,K](bf16) * W[N,K](bf16)^T  (+ epilogue)
//  EPI 0: out_f32 = acc                      (ux pass)
//  EPI 1: out_bf16 = tanh(acc + bias + add)  (DEQ iteration)
//  EPI 2: out_f32 = acc + bias               (decoder)
// Fragment layouts per CDNA5 ISA 7.12.2 (wave32).
// WG tile 256x128, 8 waves (4x2), 64x64 per wave, double-buffered LDS.
// ---------------------------------------------------------------------------
template <int EPI>
__global__ __launch_bounds__(NTHREADS)
void wmma_gemm_kernel(const bf16* __restrict__ A,
                      const bf16* __restrict__ W,
                      const float* __restrict__ bias,
                      const float* __restrict__ add,
                      void* __restrict__ out,
                      int M, int N, int K) {
  __shared__ bf16 sA[2][BM * LDSK];   // 2 x 20KB
  __shared__ bf16 sW[2][BN * LDSK];   // 2 x 10KB

  const int tid  = threadIdx.x;
  const int m0   = blockIdx.y * BM;
  const int n0   = blockIdx.x * BN;
  const int wave = tid >> 5;
  const int lane = tid & 31;
  const int wm   = wave >> 1;   // 0..3 -> 64 M rows each
  const int wn   = wave & 1;    // 0..1 -> 64 N cols each
  const int lhalf = lane >> 4;  // lane half (0/1)
  const int lmod  = lane & 15;

  // ---- branch-free staging map: v8bf chunks, 4 chunks/row (BK=32) ----
  int arow[4], acol[4];
#pragma unroll
  for (int j = 0; j < 4; ++j) {
    int c = tid + j * NTHREADS;         // 1024 chunks for A (256x32)
    arow[j] = c >> 2; acol[j] = (c & 3) * 8;
  }
  int wrow[2], wcol[2];
#pragma unroll
  for (int j = 0; j < 2; ++j) {
    int c = tid + j * NTHREADS;         // 512 chunks for W (128x32)
    wrow[j] = c >> 2; wcol[j] = (c & 3) * 8;
  }

  v8bf ra[4], rw[2];
  auto loadTile = [&](int k0) {
#pragma unroll
    for (int j = 0; j < 4; ++j)
      ra[j] = *(const v8bf*)&A[(size_t)(m0 + arow[j]) * K + k0 + acol[j]];
#pragma unroll
    for (int j = 0; j < 2; ++j)
      rw[j] = *(const v8bf*)&W[(size_t)(n0 + wrow[j]) * K + k0 + wcol[j]];
  };
  auto storeTile = [&](int b) {
#pragma unroll
    for (int j = 0; j < 4; ++j)
      *(v8bf*)&sA[b][arow[j] * LDSK + acol[j]] = ra[j];
#pragma unroll
    for (int j = 0; j < 2; ++j)
      *(v8bf*)&sW[b][wrow[j] * LDSK + wcol[j]] = rw[j];
  };

  const v8f vzero = {0.f, 0.f, 0.f, 0.f, 0.f, 0.f, 0.f, 0.f};
  v8f acc[4][4];
#pragma unroll
  for (int mf = 0; mf < 4; ++mf)
#pragma unroll
    for (int nf = 0; nf < 4; ++nf) acc[mf][nf] = vzero;

  // ---- software pipeline: stage tile 0, then overlap next-tile loads ----
  loadTile(0);
  storeTile(0);
  __syncthreads();

  int buf = 0;
  for (int k0 = 0; k0 < K; k0 += BK) {
    const bool notLast = (k0 + BK) < K;
    if (notLast) loadTile(k0 + BK);          // global_load_b128, waits deferred
    if (k0 + 2 * BK < K) {                    // gfx1250 global_prefetch_b8
      __builtin_prefetch(&A[(size_t)(m0 + arow[0]) * K + k0 + 2 * BK + acol[0]], 0, 0);
      __builtin_prefetch(&W[(size_t)(n0 + wrow[0]) * K + k0 + 2 * BK + wcol[0]], 0, 0);
    }

    // ---- preload ALL fragments (16 pipelined ds_load_b128) ----
    v16bf afrag[4], bfrag[4];
#pragma unroll
    for (int mf = 0; mf < 4; ++mf) {
      // A 16x32: lane half 0 -> K{0..7,16..23}; half 1 -> K{8..15,24..31}
      const int row = wm * 64 + mf * 16 + lmod;
      afrag[mf] = frag16(&sA[buf][row * LDSK + lhalf * 8],
                         &sA[buf][row * LDSK + 16 + lhalf * 8]);
    }
#pragma unroll
    for (int nf = 0; nf < 4; ++nf) {
      // B 32x16: lane half 0 -> K{0..7,16..23}; half 1 -> K{8..15,24..31} (col = lane%16)
      const int row = wn * 64 + nf * 16 + lmod;
      bfrag[nf] = frag16(&sW[buf][row * LDSK + lhalf * 8],
                         &sW[buf][row * LDSK + 16 + lhalf * 8]);
    }

    // ---- 16 WMMAs; only partial dscnt waits needed ----
#pragma unroll
    for (int nf = 0; nf < 4; ++nf)
#pragma unroll
      for (int mf = 0; mf < 4; ++mf)
        acc[mf][nf] = __builtin_amdgcn_wmma_f32_16x16x32_bf16(
            false, afrag[mf], false, bfrag[nf],
            (short)0, acc[mf][nf], false, false);

    if (notLast) {
      storeTile(buf ^ 1);                    // ds_store into other buffer
      __syncthreads();                       // uniform branch -> safe
      buf ^= 1;
    }
  }

  // ---- epilogue: C layout row = vgpr + 8*lhalf, col = lane%16 ----
#pragma unroll
  for (int mf = 0; mf < 4; ++mf) {
#pragma unroll
    for (int nf = 0; nf < 4; ++nf) {
      const int col   = n0 + wn * 64 + nf * 16 + lmod;
      const int rbase = m0 + wm * 64 + mf * 16 + lhalf * 8;
      const float bcol = (EPI != 0) ? bias[col] : 0.0f;
#pragma unroll
      for (int e = 0; e < 8; ++e) {
        const size_t idx = (size_t)(rbase + e) * N + col;
        const float v = acc[mf][nf][e];
        if (EPI == 0) {
          ((float*)out)[idx] = v;
        } else if (EPI == 1) {
          ((bf16*)out)[idx] = f2bf(tanhf(v + bcol + add[idx]));
        } else {
          ((float*)out)[idx] = v + bcol;
        }
      }
    }
  }
}

// ---------------------------------------------------------------------------
extern "C" void kernel_launch(void* const* d_in, const int* in_sizes, int n_in,
                              void* d_out, int out_size, void* d_ws,
                              size_t ws_size, hipStream_t stream) {
  (void)in_sizes; (void)n_in; (void)out_size; (void)ws_size;

  const float* x  = (const float*)d_in[0];   // [B, DIN]
  const float* Wx = (const float*)d_in[1];   // [H, DIN]
  const float* Wz = (const float*)d_in[2];   // [H, H]
  const float* bz = (const float*)d_in[3];   // [H]
  const float* Wd = (const float*)d_in[4];   // [DOUT, H]
  const float* bd = (const float*)d_in[5];   // [DOUT]
  float* out = (float*)d_out;                // [B, DOUT]

  const int B = 4096, DIN = 1024, H = 2048, DOUT = 1024;
  const int N_ITERS = 30;

  // ---- workspace partition (~88 MB) ----
  char* p = (char*)d_ws;
  bf16* xb  = (bf16*)p;  p += (size_t)B * DIN * 2;     //  8 MB
  bf16* Wxb = (bf16*)p;  p += (size_t)H * DIN * 2;     //  4 MB
  bf16* Wzb = (bf16*)p;  p += (size_t)H * H * 2;       //  8 MB
  bf16* Wdb = (bf16*)p;  p += (size_t)DOUT * H * 2;    //  4 MB
  float* ux = (float*)p; p += (size_t)B * H * 4;       // 32 MB
  bf16* zA  = (bf16*)p;  p += (size_t)B * H * 2;       // 16 MB
  bf16* zB  = (bf16*)p;                                // 16 MB

  const int CT = 256;
  auto blk4 = [](int n) { return (n / 4 + 255) / 256; };

  // fp32 -> bf16 conversions
  cvt_f32_bf16_kernel<<<blk4(B * DIN),  CT, 0, stream>>>(x,  xb,  B * DIN / 4);
  cvt_f32_bf16_kernel<<<blk4(H * DIN),  CT, 0, stream>>>(Wx, Wxb, H * DIN / 4);
  cvt_f32_bf16_kernel<<<blk4(H * H),    CT, 0, stream>>>(Wz, Wzb, H * H / 4);
  cvt_f32_bf16_kernel<<<blk4(DOUT * H), CT, 0, stream>>>(Wd, Wdb, DOUT * H / 4);

  // ux = x @ Wx^T  (f32 out)
  wmma_gemm_kernel<0><<<dim3(H / BN, B / BM), NTHREADS, 0, stream>>>(
      xb, Wxb, nullptr, nullptr, (void*)ux, B, H, DIN);

  // z1 = tanh(bz + ux)
  init_z_kernel<<<blk4(B * H), CT, 0, stream>>>(ux, bz, zA, H, B * H / 4);

  // iterations 2..30: z <- tanh(z @ Wz^T + bz + ux)
  bf16* zin = zA;
  bf16* zout = zB;
  for (int it = 0; it < N_ITERS - 1; ++it) {
    wmma_gemm_kernel<1><<<dim3(H / BN, B / BM), NTHREADS, 0, stream>>>(
        zin, Wzb, bz, ux, (void*)zout, B, H, H);
    bf16* t = zin; zin = zout; zout = t;
  }

  // out = z @ Wd^T + bd   (zin holds the final z)
  wmma_gemm_kernel<2><<<dim3(DOUT / BN, B / BM), NTHREADS, 0, stream>>>(
      zin, Wdb, bd, nullptr, (void*)out, B, DOUT, H);
}
